// ReturnTypeInference_7481833029895
// MI455X (gfx1250) — matrix operand order repaired and checked by reading, not compiled, not run on hardware
//
#include <hip/hip_runtime.h>
#include <hip/hip_bf16.h>
#include <math.h>

// ---------- types ----------
typedef __attribute__((ext_vector_type(16))) __bf16 v16bf;
typedef __attribute__((ext_vector_type(8)))  __bf16 v8bf;
typedef __attribute__((ext_vector_type(8)))  float  v8f;

__device__ __forceinline__ unsigned short f2bfbits(float f) {
    union { float f; unsigned u; } v; v.f = f;
    return (unsigned short)((v.u + 0x7FFFu + ((v.u >> 16) & 1u)) >> 16);  // RNE
}

__device__ __forceinline__ void atomicMaxF(float* addr, float val) {
    // ordered-int trick; addr must be initialized to -inf
    if (val >= 0.0f) atomicMax((int*)addr, __float_as_int(val));
    else             atomicMin((unsigned int*)addr, __float_as_uint(val));
}

// ---------- generic fill ----------
__global__ void k_fill_f32(float* __restrict__ p, float v, long long n) {
    long long i = (long long)blockIdx.x * blockDim.x + threadIdx.x;
    if (i < n) p[i] = v;
}

// ---------- f32 -> bf16 row-major copy ----------
__global__ void k_f32_to_bf16(const float* __restrict__ in, unsigned short* __restrict__ out,
                              long long n) {
    long long i = (long long)blockIdx.x * blockDim.x + threadIdx.x;
    if (i < n) out[i] = f2bfbits(in[i]);
}

// ---------- W[128,Ncols] f32 -> Wt[Ncols,128] bf16 (transpose) ----------
__global__ void k_w_to_bf16_T(const float* __restrict__ W, unsigned short* __restrict__ Wt,
                              int Ncols) {
    long long i = (long long)blockIdx.x * blockDim.x + threadIdx.x;
    if (i >= (long long)Ncols * 128) return;
    int c = (int)(i >> 7);
    int k = (int)(i & 127);
    Wt[(size_t)c * 128 + k] = f2bfbits(W[(size_t)k * Ncols + c]);
}

// ---------- WMMA GEMM: Y[M,Ncols] = Xb[M,128] @ Wt[Ncols,128]^T (+ bias) ----------
// Inputs are pre-converted bf16 (X row-major, W transposed), staged into LDS with
// async global->LDS b128 copies (ASYNCcnt). block = 128 threads (4 waves);
// block tile = 64 rows x 128 cols; wave tile = 16 x 128.
#define KDIM 128
#define LDA 136   // bf16 elements per LDS row; 136*2=272 bytes (16B multiple)

__global__ __launch_bounds__(128)
void k_gemm_bf16_wmma(const unsigned short* __restrict__ Xb,
                      const unsigned short* __restrict__ Wtb,
                      const float* __restrict__ bias, float* __restrict__ Y,
                      int M, int Ncols) {
    __shared__ __attribute__((aligned(16))) __bf16 Xs[64  * LDA];   // rows of X
    __shared__ __attribute__((aligned(16))) __bf16 Ws[128 * LDA];   // Ws[c][k]

    const int rowbase = blockIdx.x * 64;
    const int colbase = blockIdx.y * 128;
    const int tid = threadIdx.x;

    // --- async stage X tile: 64 rows x 128 bf16 = 64 x 16 chunks of 16B ---
    for (int i = tid; i < 64 * 16; i += 128) {
        int r = i >> 4, ck = i & 15;
        int gr = rowbase + r; if (gr >= M) gr = M - 1;        // clamp; guarded at store
        unsigned voff = ((unsigned)gr * 128u + (unsigned)ck * 8u) * 2u;
        unsigned lds  = (unsigned)(size_t)&Xs[r * LDA + ck * 8];
        asm volatile("global_load_async_to_lds_b128 %0, %1, %2 offset:0"
                     :: "v"(lds), "v"(voff), "s"(Xb) : "memory");
    }
    // --- async stage Wt chunk: 128 cols x 128 bf16 ---
    for (int i = tid; i < 128 * 16; i += 128) {
        int c = i >> 4, ck = i & 15;
        int gc = colbase + c; if (gc >= Ncols) gc = Ncols - 1; // clamp; guarded at store
        unsigned voff = ((unsigned)gc * 128u + (unsigned)ck * 8u) * 2u;
        unsigned lds  = (unsigned)(size_t)&Ws[c * LDA + ck * 8];
        asm volatile("global_load_async_to_lds_b128 %0, %1, %2 offset:0"
                     :: "v"(lds), "v"(voff), "s"(Wtb) : "memory");
    }
    asm volatile("s_wait_asynccnt 0x0" ::: "memory");
    __syncthreads();

    const int wave = tid >> 5;         // 0..3
    const int lane = tid & 31;
    const int lr   = lane & 15;
    const int hi   = lane >> 4;
    const int mrow = wave * 16;        // wave's 16-row strip inside block tile

    v8f acc[8];
    #pragma unroll
    for (int n = 0; n < 8; ++n)
        #pragma unroll
        for (int j = 0; j < 8; ++j) acc[n][j] = 0.0f;

    #pragma unroll
    for (int kk = 0; kk < 4; ++kk) {
        // A fragment: 16x32 bf16, per ISA layout (lane-hi selects K sub-block of 8, +16)
        const __bf16* arow = &Xs[(mrow + lr) * LDA + kk * 32];
        v8bf alo = *(const v8bf*)(arow + hi * 8);
        v8bf ahi = *(const v8bf*)(arow + 16 + hi * 8);
        v16bf a;
        #pragma unroll
        for (int j = 0; j < 8; ++j) { a[j] = alo[j]; a[8 + j] = ahi[j]; }

        #pragma unroll
        for (int n = 0; n < 8; ++n) {
            // B fragment: 32x16 bf16; lane col = n*16+lr, K = kk*32 + hi*8 + [0..7], +16
            const __bf16* bcol = &Ws[(n * 16 + lr) * LDA + kk * 32 + hi * 8];
            v8bf blo = *(const v8bf*)(bcol);
            v8bf bhi = *(const v8bf*)(bcol + 16);
            v16bf b;
            #pragma unroll
            for (int j = 0; j < 8; ++j) { b[j] = blo[j]; b[8 + j] = bhi[j]; }

            acc[n] = __builtin_amdgcn_wmma_f32_16x16x32_bf16(
                false, a, false, b, (short)0, acc[n], false, false);
        }
    }

    // store: C layout — lane holds col = n*16+lr, rows M = hi*8 + r
    #pragma unroll
    for (int n = 0; n < 8; ++n) {
        int col = colbase + n * 16 + lr;
        if (col >= Ncols) continue;
        float bv = bias ? bias[col] : 0.0f;
        #pragma unroll
        for (int r = 0; r < 8; ++r) {
            int grow = rowbase + mrow + hi * 8 + r;
            if (grow < M) Y[(size_t)grow * Ncols + col] = acc[n][r] + bv;
        }
    }
}

// ---------- per-node attention-coefficient dots ----------
__global__ void k_rowdots(const float* __restrict__ h,
                          const float* __restrict__ a_src, const float* __restrict__ a_dst,
                          float* __restrict__ as_, float* __restrict__ ad_, int N) {
    int n = blockIdx.x * blockDim.x + threadIdx.x;
    if (n >= N) return;
    const float* hr = h + (size_t)n * 128;
    float s0 = 0.0f, s1 = 0.0f;
    #pragma unroll 8
    for (int k = 0; k < 128; ++k) { float v = hr[k]; s0 += v * a_src[k]; s1 += v * a_dst[k]; }
    as_[n] = s0; ad_[n] = s1;
}

// ---------- edge pass 1: e = leakyrelu(as[src]+ad[dst]); segment max over dst ----------
__global__ void k_edge1(const int* __restrict__ rel, int E, int N,
                        const float* __restrict__ as_, const float* __restrict__ ad_,
                        float* __restrict__ e, float* __restrict__ m) {
    int i = blockIdx.x * blockDim.x + threadIdx.x;
    int Et = E + N;
    if (i >= Et) return;
    int src = (i < E) ? rel[i]     : (i - E);
    int dst = (i < E) ? rel[E + i] : (i - E);
    float t = as_[src] + ad_[dst];
    float ev = (t > 0.0f) ? t : 0.2f * t;
    e[i] = ev;
    atomicMaxF(&m[dst], ev);
}

// ---------- edge pass 2: ex = exp(e - m[dst]); segment sum ----------
__global__ void k_edge2(const int* __restrict__ rel, int E, int N,
                        const float* __restrict__ e, const float* __restrict__ m,
                        float* __restrict__ ex, float* __restrict__ s) {
    int i = blockIdx.x * blockDim.x + threadIdx.x;
    int Et = E + N;
    if (i >= Et) return;
    int dst = (i < E) ? rel[E + i] : (i - E);
    float v = __expf(e[i] - m[dst]);
    ex[i] = v;
    atomicAdd(&s[dst], v);
}

// ---------- edge pass 3: agg[dst] += (ex/s[dst]) * h[src]  (one thread per edge-col) ----------
__global__ void k_edge3(const int* __restrict__ rel, int E, int N,
                        const float* __restrict__ ex, const float* __restrict__ s,
                        const float* __restrict__ h, float* __restrict__ agg) {
    long long idx = (long long)blockIdx.x * blockDim.x + threadIdx.x;
    long long Et = (long long)E + N;
    if (idx >= Et * 128) return;
    int i = (int)(idx >> 7);
    int c = (int)(idx & 127);
    int src = (i < E) ? rel[i]     : (i - E);
    int dst = (i < E) ? rel[E + i] : (i - E);
    float alpha = ex[i] / s[dst];
    atomicAdd(&agg[(size_t)dst * 128 + c], alpha * h[(size_t)src * 128 + c]);
}

// ---------- bias add (+ optional ELU) ----------
__global__ void k_bias_act(const float* __restrict__ in, const float* __restrict__ b,
                           float* __restrict__ out, long long total, int doElu) {
    long long i = (long long)blockIdx.x * blockDim.x + threadIdx.x;
    if (i >= total) return;
    float v = in[i] + b[i & 127];
    if (doElu) v = (v > 0.0f) ? v : (__expf(v) - 1.0f);
    out[i] = v;
}

// ---------- per-node energy: relu(p + bp1) . Wp2 + bp2 ----------
__global__ void k_energy(const float* __restrict__ p, const float* __restrict__ bp1,
                         const float* __restrict__ Wp2, const float* __restrict__ bp2,
                         float* __restrict__ energy, int N, int HP) {
    int n = blockIdx.x * blockDim.x + threadIdx.x;
    if (n >= N) return;
    float e = bp2[0];
    const float* pr = p + (size_t)n * HP;
    for (int j = 0; j < HP; ++j) {
        float t = pr[j] + bp1[j];
        t = (t > 0.0f) ? t : 0.0f;
        e += t * Wp2[j];
    }
    energy[n] = e;
}

// ---------- segment id (searchsorted-right) + segment max of energy ----------
__global__ void k_seg(const int* __restrict__ node_pos, const float* __restrict__ energy,
                      int* __restrict__ seg, float* __restrict__ segm, int N, int G) {
    int n = blockIdx.x * blockDim.x + threadIdx.x;
    if (n >= N) return;
    int lo = 0, hh = G + 1;
    while (lo < hh) { int mid = (lo + hh) >> 1; if (node_pos[mid] <= n) lo = mid + 1; else hh = mid; }
    int g = lo - 1;
    g = (g < 0) ? 0 : ((g > G - 1) ? (G - 1) : g);
    seg[n] = g;
    atomicMaxF(&segm[g], energy[n]);
}

// ---------- segment softmax numerator + denominator ----------
__global__ void k_segexp(const float* __restrict__ energy, const int* __restrict__ seg,
                         const float* __restrict__ segm, float* __restrict__ segs,
                         float* __restrict__ exn, int N) {
    int n = blockIdx.x * blockDim.x + threadIdx.x;
    if (n >= N) return;
    int g = seg[n];
    float v = __expf(energy[n] - segm[g]);
    exn[n] = v;
    atomicAdd(&segs[g], v);
}

// ---------- weighted pooling scatter: pooled[g] += w_n * x[n] ----------
__global__ void k_pool(const float* __restrict__ x, const float* __restrict__ exn,
                       const float* __restrict__ segs, const int* __restrict__ seg,
                       float* __restrict__ pooled, int N) {
    long long idx = (long long)blockIdx.x * blockDim.x + threadIdx.x;
    if (idx >= (long long)N * 128) return;
    int n = (int)(idx >> 7);
    int c = (int)(idx & 127);
    int g = seg[n];
    float w = exn[n] / segs[g];
    atomicAdd(&pooled[(size_t)g * 128 + c], w * x[idx]);
}

// ================== host-side orchestration ==================
static inline int cdiv(long long a, long long b) { return (int)((a + b - 1) / b); }

extern "C" void kernel_launch(void* const* d_in, const int* in_sizes, int n_in,
                              void* d_out, int out_size, void* d_ws, size_t ws_size,
                              hipStream_t stream) {
    const float* cfg_nodes = (const float*)d_in[0];
    const int*   rel       = (const int*)  d_in[1];
    const int*   node_pos  = (const int*)  d_in[2];
    const float* W_g1      = (const float*)d_in[3];
    const float* att_src1  = (const float*)d_in[4];
    const float* att_dst1  = (const float*)d_in[5];
    const float* b_g1      = (const float*)d_in[6];
    const float* W_g2      = (const float*)d_in[7];
    const float* att_src2  = (const float*)d_in[8];
    const float* att_dst2  = (const float*)d_in[9];
    const float* b_g2      = (const float*)d_in[10];
    const float* Wp1       = (const float*)d_in[11];
    const float* bp1       = (const float*)d_in[12];
    const float* Wp2       = (const float*)d_in[13];
    const float* bp2       = (const float*)d_in[14];
    const float* Wl        = (const float*)d_in[15];
    const float* bl        = (const float*)d_in[16];

    const int N  = in_sizes[0] / 128;
    const int E  = in_sizes[1] / 2;
    const int G  = in_sizes[2] - 1;
    const int HP = in_sizes[12];
    const int C  = in_sizes[16];
    const long long Et = (long long)E + N;

    // workspace carve-out
    float* ws     = (float*)d_ws;
    float* h      = ws;                       // N*128 (also reused as x2 after layer 2)
    float* agg    = h      + (size_t)N * 128;
    float* xcur   = agg    + (size_t)N * 128;
    float* as_    = xcur   + (size_t)N * 128;
    float* ad_    = as_    + N;
    float* mbuf   = ad_    + N;
    float* sbuf   = mbuf   + N;
    float* ebuf   = sbuf   + N;
    float* exbuf  = ebuf   + Et;
    float* pbuf   = exbuf  + Et;              // N*HP
    float* energy = pbuf   + (size_t)N * HP;
    float* exn    = energy + N;
    int*   seg    = (int*)(exn + N);
    float* segm   = (float*)(seg + N);
    float* segs   = segm + G;
    float* pooled = segs + G;                 // G*128
    unsigned short* Xbuf  = (unsigned short*)(pooled + (size_t)G * 128);  // N*128 bf16
    unsigned short* Wtbuf = Xbuf + (size_t)N * 128;                        // 1024*128 bf16

    const int T = 256;
    dim3 blk(T);

    auto gemm = [&](const float* X, const float* W, const float* bias,
                    float* Y, int M, int Ncols) {
        k_f32_to_bf16<<<cdiv((long long)M * 128, T), blk, 0, stream>>>(X, Xbuf, (long long)M * 128);
        k_w_to_bf16_T<<<cdiv((long long)Ncols * 128, T), blk, 0, stream>>>(W, Wtbuf, Ncols);
        k_gemm_bf16_wmma<<<dim3(cdiv(M, 64), cdiv(Ncols, 128)), dim3(128), 0, stream>>>(
            Xbuf, Wtbuf, bias, Y, M, Ncols);
    };

    auto run_layer = [&](const float* x_in, const float* W,
                         const float* a_s, const float* a_d) {
        // h = x_in @ W   (N x 128) via WMMA
        gemm(x_in, W, nullptr, h, N, 128);
        // attention coefficients
        k_rowdots<<<cdiv(N, T), blk, 0, stream>>>(h, a_s, a_d, as_, ad_, N);
        // reset segment buffers + accumulator
        k_fill_f32<<<cdiv(N, T), blk, 0, stream>>>(mbuf, -INFINITY, N);
        k_fill_f32<<<cdiv(N, T), blk, 0, stream>>>(sbuf, 0.0f, N);
        k_fill_f32<<<cdiv((long long)N * 128, T), blk, 0, stream>>>(agg, 0.0f, (long long)N * 128);
        // edge softmax + aggregate
        k_edge1<<<cdiv(Et, T), blk, 0, stream>>>(rel, E, N, as_, ad_, ebuf, mbuf);
        k_edge2<<<cdiv(Et, T), blk, 0, stream>>>(rel, E, N, ebuf, mbuf, exbuf, sbuf);
        k_edge3<<<cdiv(Et * 128, T), blk, 0, stream>>>(rel, E, N, exbuf, sbuf, h, agg);
    };

    // ----- layer 1 -----
    run_layer(cfg_nodes, W_g1, att_src1, att_dst1);
    k_bias_act<<<cdiv((long long)N * 128, T), blk, 0, stream>>>(agg, b_g1, xcur, (long long)N * 128, 1);

    // ----- layer 2 -----
    run_layer(xcur, W_g2, att_src2, att_dst2);
    // x2 stored into h (its GEMM contents are no longer needed)
    k_bias_act<<<cdiv((long long)N * 128, T), blk, 0, stream>>>(agg, b_g2, h, (long long)N * 128, 0);
    const float* x2 = h;

    // ----- pooling attention -----
    gemm(x2, Wp1, nullptr, pbuf, N, HP);      // p = x2 @ Wp1 (cols clamped inside kernel)
    k_energy<<<cdiv(N, T), blk, 0, stream>>>(pbuf, bp1, Wp2, bp2, energy, N, HP);

    k_fill_f32<<<cdiv(G, T), blk, 0, stream>>>(segm, -INFINITY, G);
    k_fill_f32<<<cdiv(G, T), blk, 0, stream>>>(segs, 0.0f, G);
    k_fill_f32<<<cdiv((long long)G * 128, T), blk, 0, stream>>>(pooled, 0.0f, (long long)G * 128);

    k_seg<<<cdiv(N, T), blk, 0, stream>>>(node_pos, energy, seg, segm, N, G);
    k_segexp<<<cdiv(N, T), blk, 0, stream>>>(energy, seg, segm, segs, exn, N);
    k_pool<<<cdiv((long long)N * 128, T), blk, 0, stream>>>(x2, exn, segs, seg, pooled, N);

    // ----- classifier: out = pooled @ Wl + bl   (G x C) via WMMA -----
    gemm(pooled, Wl, bl, (float*)d_out, G, C);
}
